// DiffeqZeroTraceAttention_51178830299528
// MI455X (gfx1250) — compile-verified, matrix operand order checked by reading, not run on hardware
//
#include <hip/hip_runtime.h>
#include <hip/hip_bf16.h>
#include <math.h>

// ---------------------------------------------------------------------------
// DiffeqZeroTraceAttention for MI455X (gfx1250, wave32, WMMA f16->f32)
// B=2, N=1024, D=8, h=128, Hid=256, H=4 heads, dh=32
// ---------------------------------------------------------------------------

typedef __attribute__((ext_vector_type(16))) _Float16 v16h;
typedef __attribute__((ext_vector_type(8)))  _Float16 v8h;
typedef __attribute__((ext_vector_type(8)))  float    v8f;

#define Bb   2
#define Nn   1024
#define Dd   8
#define Hh   128
#define HID  256
#define NH   4
#define DH   32

// WMMA wrapper: D = A(16x32 f16) * B(32x16 f16) + C(16x16 f32)
__device__ __forceinline__ v8f wmma_f16(v16h a, v16h b, v8f c) {
  return __builtin_amdgcn_wmma_f32_16x16x32_f16(
      /*neg_a=*/false, a, /*neg_b=*/false, b,
      /*c_mod=*/(short)0, c, /*reuse_a=*/false, /*reuse_b=*/false);
}

// Branchless fast tanh: 1 - 2/(exp(2x)+1).  exp overflow -> rcp(inf)=0 -> +/-1.
__device__ __forceinline__ float fast_tanh(float x) {
  float e = __expf(2.0f * x);
  return 1.0f - 2.0f * __builtin_amdgcn_rcpf(e + 1.0f);
}

// Load one lane's A/B-matrix fragment (16x32 f16) from a 16B-aligned row of 32
// contiguous halves.  Layout per CDNA5 ISA 7.12.2:
//   lane = m + 16*((k>>3)&1), element e = (k&7) + 8*(k>>4)
// Two b128 loads per lane.
__device__ __forceinline__ v16h load_row32(const _Float16* __restrict__ rp, int lg) {
  v8h lo = *(const v8h*)(rp + 8 * lg);        // k = 8*lg .. 8*lg+7
  v8h hi = *(const v8h*)(rp + 16 + 8 * lg);   // k = 16+8*lg .. 16+8*lg+7
  v16h a;
#pragma unroll
  for (int e = 0; e < 8; ++e) { a[e] = lo[e]; a[e + 8] = hi[e]; }
  return a;
}

// ---------------------------------------------------------------------------
// Fused 3-layer MADE MLP.  One workgroup = 16 tokens, 256 threads (8 waves).
//   L1: (16x8)   @ W0(8x256)   -> tanh -> LDS h1   (scalar, K=8)
//   L2: (16x256) @ W1(256x256) -> tanh -> LDS h2   (WMMA, W staged transposed)
//   L3: (16x256) @ W2(256xoutw)         -> global  (WMMA, W staged transposed)
// qmode!=0 applies the MADE masks (L1 i<=j%7, L2 i%7<=j%7, L3 i%7<j%8) and
// scatters output as [b][d][n][h] (query layout). Else [b][n][h].
// ---------------------------------------------------------------------------
__global__ __launch_bounds__(256)
void made_mlp_kernel(const float* __restrict__ x,
                     const float* __restrict__ W0, const float* __restrict__ bias0,
                     const float* __restrict__ W1, const float* __restrict__ bias1,
                     const float* __restrict__ W2, const float* __restrict__ bias2,
                     _Float16* __restrict__ out, int outw, int qmode)
{
  __shared__ __align__(16) _Float16 h1[16 * HID];
  __shared__ __align__(16) _Float16 h2[16 * HID];
  __shared__ __align__(16) _Float16 wbuf[256 * 32];  // transposed [col][k] chunk
  __shared__ __align__(16) float    xs[16 * Dd];

  const int blk  = blockIdx.x;
  const int b    = blk >> 6;            // / (N/16)
  const int n0   = (blk & 63) << 4;     // token tile base
  const int t    = threadIdx.x;
  const int lane = t & 31;
  const int wave = t >> 5;
  const int l16  = lane & 15;
  const int lg   = lane >> 4;

  if (t < 16 * Dd) xs[t] = x[(size_t)(b * Nn + n0) * Dd + t];
  __syncthreads();

  // ---- Layer 1: K=8, scalar VALU (one column per thread) ----
  {
    const int j  = t;
    const int jm = j % 7;
    float w[8];
#pragma unroll
    for (int i = 0; i < 8; ++i) {
      float wv = W0[i * HID + j];
      if (qmode && !(i <= jm)) wv = 0.f;
      w[i] = wv;
    }
    const float bj = bias0[j];
#pragma unroll 4
    for (int r = 0; r < 16; ++r) {
      float s = bj;
#pragma unroll
      for (int i = 0; i < 8; ++i) s += xs[r * 8 + i] * w[i];
      h1[r * HID + j] = (_Float16)fast_tanh(s);
    }
  }
  __syncthreads();

  // ---- Layer 2: h1 @ W1 (WMMA). Each wave owns 2 column tiles. ----
  {
    const int c0  = wave * 32;
    const int jm7 = t % 7;
    v8f acc0 = {}, acc1 = {};
    for (int k0 = 0; k0 < HID; k0 += 32) {
      __syncthreads();  // protect wbuf from previous iteration's readers
      // stage transposed chunk: thread t owns column t, 4x b128 LDS stores
#pragma unroll
      for (int part = 0; part < 4; ++part) {
        v8h tmp;
#pragma unroll
        for (int e = 0; e < 8; ++e) {
          int k = k0 + part * 8 + e;
          float wv = W1[(size_t)k * HID + t];
          if (qmode && !((k % 7) <= jm7)) wv = 0.f;
          tmp[e] = (_Float16)wv;
        }
        *(v8h*)&wbuf[t * 32 + part * 8] = tmp;
      }
      __syncthreads();

      v16h a   = load_row32(&h1[l16 * HID + k0], lg);
      v16h bv0 = load_row32(&wbuf[(c0 + l16) * 32], lg);
      v16h bv1 = load_row32(&wbuf[(c0 + 16 + l16) * 32], lg);
      acc0 = wmma_f16(a, bv0, acc0);
      acc1 = wmma_f16(a, bv1, acc1);
    }
    // bias + tanh -> h2 (each wave writes its own columns)
#pragma unroll
    for (int r = 0; r < 8; ++r) {
      int m = r + 8 * lg;
      h2[m * HID + c0 + l16]      = (_Float16)fast_tanh(acc0[r] + bias1[c0 + l16]);
      h2[m * HID + c0 + 16 + l16] = (_Float16)fast_tanh(acc1[r] + bias1[c0 + 16 + l16]);
    }
  }
  __syncthreads();

  // ---- Layer 3: h2 @ W2 (WMMA, weights staged transposed per 256-col group) --
  for (int cg = 0; cg < outw; cg += 256) {
    const int cols   = (outw - cg) < 256 ? (outw - cg) : 256;  // 256 or 128
    const int ntiles = cols >> 7;                              // 2 or 1
    const int c0     = wave * 16 * ntiles;
    v8f acc[2];
    for (int tt = 0; tt < ntiles; ++tt) {
      const float bj = bias2[cg + c0 + tt * 16 + l16];
#pragma unroll
      for (int r = 0; r < 8; ++r) acc[tt][r] = bj;
    }
    for (int k0 = 0; k0 < HID; k0 += 32) {
      __syncthreads();
      if (t < cols) {
        const int j  = cg + t;
        const int jd = j & 7;
#pragma unroll
        for (int part = 0; part < 4; ++part) {
          v8h tmp;
#pragma unroll
          for (int e = 0; e < 8; ++e) {
            int k = k0 + part * 8 + e;
            float wv = W2[(size_t)k * outw + j];
            if (qmode && !((k % 7) < jd)) wv = 0.f;  // strict < on output mask
            tmp[e] = (_Float16)wv;
          }
          *(v8h*)&wbuf[t * 32 + part * 8] = tmp;
        }
      }
      __syncthreads();
      v16h a = load_row32(&h2[l16 * HID + k0], lg);
      for (int tt = 0; tt < ntiles; ++tt) {
        v16h bv = load_row32(&wbuf[(c0 + tt * 16 + l16) * 32], lg);
        acc[tt] = wmma_f16(a, bv, acc[tt]);
      }
    }
    // store f16
    for (int tt = 0; tt < ntiles; ++tt) {
      const int j = cg + c0 + tt * 16 + l16;
#pragma unroll
      for (int r = 0; r < 8; ++r) {
        int m = r + 8 * lg;
        int n = n0 + m;
        if (qmode) {
          int d = j & 7, c = j >> 3;  // q[b,n,c*D+d] -> qf[b][d][n][c]
          out[((size_t)(b * Dd + d) * Nn + n) * Hh + c] = (_Float16)acc[tt][r];
        } else {
          out[((size_t)(b * Nn + n)) * Hh + j] = (_Float16)acc[tt][r];
        }
      }
    }
  }
}

// ---------------------------------------------------------------------------
// Flash-style attention: one wave per (b, d, head, 16-query tile).
// Per 32-key chunk: V chunk staged transposed in LDS (contiguous b128 global
// reads), 2 WMMAs for S, online softmax (shfl_xor width-16 matches the
// C-matrix half-wave row layout), P bounced through LDS to A-layout,
// 2 WMMAs for P@V with rescaled f32 accumulators. Diagonal masked out.
// ---------------------------------------------------------------------------
__global__ __launch_bounds__(32)
void zt_attn_kernel(const _Float16* __restrict__ qf,  // [B][D][N][Hh]
                    const _Float16* __restrict__ kf,  // [B][N][Hh]
                    const _Float16* __restrict__ vf,  // [B][N][Hh]
                    float* __restrict__ yout)         // [B][N][D][Hh]
{
  __shared__ __align__(16) _Float16 pbuf[16 * 32];
  __shared__ __align__(16) _Float16 vt[32 * 32];  // transposed V chunk [dh][key]

  const int blk  = blockIdx.x;
  const int qt   = blk & 63;
  const int head = (blk >> 6) & 3;
  const int d    = (blk >> 8) & 7;
  const int b    = (blk >> 11) & 1;
  const int lane = threadIdx.x;
  const int l16  = lane & 15;
  const int lg   = lane >> 4;

  const v16h aq = load_row32(
      qf + ((size_t)(b * Dd + d) * Nn + qt * 16 + l16) * Hh + head * DH, lg);

  const float scale = 0.17677669529663687f;  // 1/sqrt(32)
  float mrow[8], lrow[8];
  v8f y0 = {}, y1 = {};
#pragma unroll
  for (int r = 0; r < 8; ++r) { mrow[r] = -1e30f; lrow[r] = 0.f; }

  for (int kb = 0; kb < Nn; kb += 32) {
    // stage V chunk transposed: lane owns key kb+lane (4x b128 global loads)
    {
      const _Float16* vp = vf + ((size_t)b * Nn + kb + lane) * Hh + head * DH;
      v8h r0 = *(const v8h*)(vp);
      v8h r1 = *(const v8h*)(vp + 8);
      v8h r2 = *(const v8h*)(vp + 16);
      v8h r3 = *(const v8h*)(vp + 24);
#pragma unroll
      for (int e = 0; e < 8; ++e) {
        vt[e * 32 + lane]        = r0[e];
        vt[(8 + e) * 32 + lane]  = r1[e];
        vt[(16 + e) * 32 + lane] = r2[e];
        vt[(24 + e) * 32 + lane] = r3[e];
      }
    }

    // K^T chunk as two B tiles (keys kb..kb+15, kb+16..kb+31)
    const _Float16* kp = kf + ((size_t)b * Nn + kb + l16) * Hh + head * DH;
    v16h bk0 = load_row32(kp, lg);
    v16h bk1 = load_row32(kp + 16 * Hh, lg);
    v8f s0 = {}, s1 = {};
    s0 = wmma_f16(aq, bk0, s0);
    s1 = wmma_f16(aq, bk1, s1);

    // online softmax; row m = r + 8*lg lives across one 16-lane group
#pragma unroll
    for (int r = 0; r < 8; ++r) {
      const int m    = r + 8 * lg;
      const int qrow = qt * 16 + m;
      float v0 = s0[r] * scale;
      float v1 = s1[r] * scale;
      if (kb + l16 == qrow)      v0 = -1e30f;  // mask diagonal
      if (kb + 16 + l16 == qrow) v1 = -1e30f;
      float mx = fmaxf(v0, v1);
#pragma unroll
      for (int off = 8; off >= 1; off >>= 1) mx = fmaxf(mx, __shfl_xor(mx, off, 16));
      const float mnew  = fmaxf(mrow[r], mx);
      const float alpha = __expf(mrow[r] - mnew);
      mrow[r] = mnew;
      const float p0 = __expf(v0 - mnew);
      const float p1 = __expf(v1 - mnew);
      float rs = p0 + p1;
#pragma unroll
      for (int off = 8; off >= 1; off >>= 1) rs += __shfl_xor(rs, off, 16);
      lrow[r] = lrow[r] * alpha + rs;
      y0[r] *= alpha;
      y1[r] *= alpha;
      pbuf[m * 32 + l16]      = (_Float16)p0;   // C-layout -> row-major LDS
      pbuf[m * 32 + 16 + l16] = (_Float16)p1;
    }
    __syncthreads();  // LDS store -> load ordering (single wave)

    v16h ap  = load_row32(&pbuf[l16 * 32], lg);
    v16h bv0 = load_row32(&vt[l16 * 32], lg);         // dh cols 0..15
    v16h bv1 = load_row32(&vt[(16 + l16) * 32], lg);  // dh cols 16..31
    y0 = wmma_f16(ap, bv0, y0);
    y1 = wmma_f16(ap, bv1, y1);
    __syncthreads();  // protect pbuf/vt before next iteration
  }

  // normalize and store y[b][n][d][head*32 + c]
#pragma unroll
  for (int r = 0; r < 8; ++r) {
    const int m = r + 8 * lg;
    const float inv = __builtin_amdgcn_rcpf(lrow[r]);
    size_t base = (((size_t)b * Nn + qt * 16 + m) * Dd + d) * Hh + head * DH;
    yout[base + l16]      = y0[r] * inv;
    yout[base + 16 + l16] = y1[r] * inv;
  }
}

// ---------------------------------------------------------------------------
// Projection: out[b,n,d] = y[b,n,d,:] . pW + pb ; second output = zeros(B,N,D)
// ---------------------------------------------------------------------------
__global__ __launch_bounds__(256)
void proj_kernel(const float* __restrict__ y, const float* __restrict__ pW,
                 const float* __restrict__ pb, float* __restrict__ out)
{
  const int i = blockIdx.x * blockDim.x + threadIdx.x;
  const int total = Bb * Nn * Dd;
  if (i >= total) return;
  const float* yr = y + (size_t)i * Hh;
  float s = pb[0];
#pragma unroll 8
  for (int c = 0; c < Hh; ++c) s += yr[c] * pW[c];
  out[i] = s;
  out[total + i] = 0.f;  // zero log-det-jacobian output
}

// ---------------------------------------------------------------------------
extern "C" void kernel_launch(void* const* d_in, const int* in_sizes, int n_in,
                              void* d_out, int out_size, void* d_ws, size_t ws_size,
                              hipStream_t stream) {
  (void)in_sizes; (void)n_in; (void)out_size; (void)ws_size;
  const float* x   = (const float*)d_in[0];
  // d_in[1] = t (unused)
  const float* qW0 = (const float*)d_in[2];  const float* qb0 = (const float*)d_in[3];
  const float* qW1 = (const float*)d_in[4];  const float* qb1 = (const float*)d_in[5];
  const float* qW2 = (const float*)d_in[6];  const float* qb2 = (const float*)d_in[7];
  const float* kW0 = (const float*)d_in[8];  const float* kb0 = (const float*)d_in[9];
  const float* kW1 = (const float*)d_in[10]; const float* kb1 = (const float*)d_in[11];
  const float* kW2 = (const float*)d_in[12]; const float* kb2 = (const float*)d_in[13];
  const float* vW0 = (const float*)d_in[14]; const float* vb0 = (const float*)d_in[15];
  const float* vW1 = (const float*)d_in[16]; const float* vb1 = (const float*)d_in[17];
  const float* vW2 = (const float*)d_in[18]; const float* vb2 = (const float*)d_in[19];
  const float* pW  = (const float*)d_in[20]; const float* pb  = (const float*)d_in[21];
  float* out = (float*)d_out;

  // workspace carve-out (~13 MB)
  _Float16* qf = (_Float16*)d_ws;                       // [B][D][N][Hh] f16
  _Float16* kf = qf + (size_t)Bb * Dd * Nn * Hh;        // [B][N][Hh]    f16
  _Float16* vf = kf + (size_t)Bb * Nn * Hh;             // [B][N][Hh]    f16
  float*    yw = (float*)(vf + (size_t)Bb * Nn * Hh);   // [B][N][D][Hh] f32

  const int mlpBlocks = Bb * (Nn / 16);  // 128
  made_mlp_kernel<<<mlpBlocks, 256, 0, stream>>>(x, qW0, qb0, qW1, qb1, qW2, qb2,
                                                 qf, Hh * Dd, /*qmode=*/1);
  made_mlp_kernel<<<mlpBlocks, 256, 0, stream>>>(x, kW0, kb0, kW1, kb1, kW2, kb2,
                                                 kf, Hh, /*qmode=*/0);
  made_mlp_kernel<<<mlpBlocks, 256, 0, stream>>>(x, vW0, vb0, vW1, vb1, vW2, vb2,
                                                 vf, Hh, /*qmode=*/0);

  zt_attn_kernel<<<Bb * Dd * NH * (Nn / 16), 32, 0, stream>>>(qf, kf, vf, yw);

  const int projN = Bb * Nn * Dd;
  proj_kernel<<<(projN + 255) / 256, 256, 0, stream>>>(yw, pW, pb, out);
}
